// SNNQUT_filter_bank_91164975825529
// MI455X (gfx1250) — matrix operand order repaired and checked by reading, not compiled, not run on hardware
//
#include <hip/hip_runtime.h>

typedef __attribute__((ext_vector_type(16))) _Float16     v16h;
typedef __attribute__((ext_vector_type(8)))  float        v8f;
typedef __attribute__((ext_vector_type(8)))  unsigned int v8u;

// ---------------------------------------------------------------------------
// fake_quant forward value: clamp then round to 16 levels (round-half-even,
// matching jnp.round). STE means forward value IS the quantized value.
// ---------------------------------------------------------------------------
__device__ __forceinline__ float fq(float w, float wmin, float wmax) {
    const float inv_levels = 1.0f / 15.0f;                 // 2^4 - 1 levels
    float scale = (wmax - wmin) * inv_levels;
    float wc = fminf(fmaxf(w, wmin), wmax);
    return rintf((wc - wmin) / scale) * scale + wmin;
}

// ---------------------------------------------------------------------------
// Pack fake-quantized weights into CDNA5 WMMA B-fragment order (16x16x32 f16).
// Dword index = ((n_tile*Kt + k_tile)*32 + lane)*8 + vgpr
//   n = n_tile*16 + (lane&15), half = lane>>4
//   K pair = k_tile*32 + half*16 + 2*vgpr (+1)
// Rows >= Nrows (padding for OUT=35 -> 48) are zero.
// ---------------------------------------------------------------------------
__global__ void pack_weights_kernel(const float* __restrict__ W,
                                    unsigned int* __restrict__ out,
                                    int Nrows, int K, int Npad,
                                    float wmin, float wmax, int total) {
    int tid = blockIdx.x * blockDim.x + threadIdx.x;
    if (tid >= total) return;
    int vgpr = tid & 7;
    int lane = (tid >> 3) & 31;
    int tile = tid >> 8;
    int Kt = K >> 5;
    int k_tile = tile % Kt;
    int n_tile = tile / Kt;
    int n = n_tile * 16 + (lane & 15);
    int half = lane >> 4;
    int k0 = k_tile * 32 + half * 16 + 2 * vgpr;
    float w0 = 0.0f, w1 = 0.0f;
    if (n < Nrows) {
        const float* row = W + (size_t)n * K;
        w0 = fq(row[k0], wmin, wmax);
        w1 = fq(row[k0 + 1], wmin, wmax);
    }
    _Float16 h0 = (_Float16)w0, h1 = (_Float16)w1;
    unsigned short u0 = __builtin_bit_cast(unsigned short, h0);
    unsigned short u1 = __builtin_bit_cast(unsigned short, h1);
    out[tid] = (unsigned int)u0 | ((unsigned int)u1 << 16);
}

// x: [B, T, IN] f32  ->  xh: [T, B, IN] f16, scaled by 1/15
__global__ void convert_x_kernel(const float* __restrict__ x,
                                 _Float16* __restrict__ xh,
                                 int T, int B, int IN) {
    int tid = blockIdx.x * blockDim.x + threadIdx.x;
    int total = T * B * IN;
    if (tid >= total) return;
    int k = tid % IN;
    int r = tid / IN;
    int b = r % B;
    int t = r / B;
    xh[tid] = (_Float16)(x[((size_t)b * T + t) * IN + k] * (1.0f / 15.0f));
}

__global__ void zero_f32_kernel(float* __restrict__ p, size_t n) {
    size_t tid = (size_t)blockIdx.x * blockDim.x + threadIdx.x;
    if (tid < n) p[tid] = 0.0f;
}

// ---------------------------------------------------------------------------
// One LIF layer for one timestep: cur = act_in @ Wq^T via WMMA f16->f32,
// then membrane update + spike. One wave per 16x16 output tile.
//   act_in : [B, K]   f16 row-major (spikes or x/15)
//   wpack  : B-fragments, (Npad/16)*(K/32) tiles
//   mem    : [B, Npad] f32 persistent membrane
//   spk_out: [B, Nout] f32 slice of d_out      (n < Nout only)
//   mem_out: [B, Nout] f32 slice of d_out or null (layer 4 mem4)
//   act_out: [B, Npad] f16 spikes for next layer, or null
// ---------------------------------------------------------------------------
__global__ void __launch_bounds__(256)
snn_layer_kernel(const _Float16* __restrict__ act_in,
                 const unsigned int* __restrict__ wpack,
                 float* __restrict__ mem,
                 float* __restrict__ spk_out,
                 float* __restrict__ mem_out,
                 _Float16* __restrict__ act_out,
                 int K, int Npad, int Nout,
                 float beta, float thr, int reset_zero) {
    const int lane = threadIdx.x & 31;
    const int wave = threadIdx.x >> 5;
    const int Nt = Npad >> 4;
    const int Kt = K >> 5;
    const int tile = blockIdx.x * (blockDim.x >> 5) + wave;
    const int n_tile = tile % Nt;
    const int m_tile = tile / Nt;

    const int mrow = lane & 15;   // A row within tile (both lane halves)
    const int half = lane >> 4;   // which K-half this lane carries

    v8f acc = {};
    const _Float16* arow = act_in + (size_t)(m_tile * 16 + mrow) * K;
    const v8u* bbase = (const v8u*)wpack + (size_t)(n_tile * Kt) * 32 + lane;

    for (int kt = 0; kt < Kt; ++kt) {
        // A fragment: contiguous K order (K = half*16 + s), matching packed B
        const _Float16* ab = arow + kt * 32;
        v8u a_u;
#pragma unroll
        for (int i = 0; i < 8; ++i) {
            int off = half * 16 + 2 * i;
            a_u[i] = *(const unsigned int*)(ab + off);
        }
        // B fragment: prepacked, 8 consecutive dwords per lane (2x b128)
        v8u b_u = bbase[(size_t)kt * 32];

        v16h a = __builtin_bit_cast(v16h, a_u);
        v16h b = __builtin_bit_cast(v16h, b_u);
        acc = __builtin_amdgcn_wmma_f32_16x16x32_f16(
            false, a, false, b, (short)0, acc, false, false);
    }

    // LIF epilogue. C/D layout: VGPR r -> (M = r + 8*half, N = lane&15)
    const int n = n_tile * 16 + mrow;
#pragma unroll
    for (int r = 0; r < 8; ++r) {
        int brow = m_tile * 16 + r + half * 8;
        size_t midx = (size_t)brow * Npad + n;
        float cur   = acc[r];
        float mprev = mem[midx];
        float reset = (mprev > thr) ? 1.0f : 0.0f;   // detached, prev-mem reset
        float base  = beta * mprev + cur;
        float mnew  = reset_zero ? (base - reset * base)   // "zero"
                                 : (base - reset * thr);   // "subtract"
        mem[midx] = mnew;
        float spk = ((mnew - thr) > 0.0f) ? 1.0f : 0.0f;   // Heaviside fwd
        if (act_out) act_out[midx] = (_Float16)spk;
        if (n < Nout) {
            size_t oidx = (size_t)brow * Nout + n;
            spk_out[oidx] = spk;
            if (mem_out) mem_out[oidx] = mnew;
        }
    }
}

// ---------------------------------------------------------------------------
extern "C" void kernel_launch(void* const* d_in, const int* in_sizes, int n_in,
                              void* d_out, int out_size, void* d_ws, size_t ws_size,
                              hipStream_t stream) {
    constexpr int B = 256, T = 100, IN = 128, H = 1024, OUT = 35, OUTP = 48;

    const float* x  = (const float*)d_in[0];
    const float* w1 = (const float*)d_in[1];
    const float* w2 = (const float*)d_in[2];
    const float* w3 = (const float*)d_in[3];
    const float* w4 = (const float*)d_in[4];
    float* out = (float*)d_out;

    // ---- workspace carve-up (256B aligned) ----
    char* ws = (char*)d_ws;
    size_t off = 0;
    auto salloc = [&](size_t bytes) -> void* {
        off = (off + 255) & ~(size_t)255;
        void* p = ws + off;
        off += bytes;
        return p;
    };
    _Float16* xh      = (_Float16*)salloc((size_t)T * B * IN * 2);
    unsigned int* w1p = (unsigned int*)salloc((size_t)(H / 16) * (IN / 32) * 256 * 4);
    unsigned int* w2p = (unsigned int*)salloc((size_t)(H / 16) * (H / 32) * 256 * 4);
    unsigned int* w3p = (unsigned int*)salloc((size_t)(H / 16) * (H / 32) * 256 * 4);
    unsigned int* w4p = (unsigned int*)salloc((size_t)(OUTP / 16) * (H / 32) * 256 * 4);
    float* mems = (float*)salloc(((size_t)3 * B * H + (size_t)B * OUTP) * 4);
    float* mem1 = mems;
    float* mem2 = mems + (size_t)B * H;
    float* mem3 = mems + (size_t)2 * B * H;
    float* mem4 = mems + (size_t)3 * B * H;
    _Float16* act1 = (_Float16*)salloc((size_t)B * H * 2);
    _Float16* act2 = (_Float16*)salloc((size_t)B * H * 2);
    _Float16* act3 = (_Float16*)salloc((size_t)B * H * 2);

    // ---- quantize + pack weights into WMMA B-fragment layout ----
    {
        int tot = (H / 16) * (IN / 32) * 256;
        pack_weights_kernel<<<(tot + 255) / 256, 256, 0, stream>>>(
            w1, w1p, H, IN, H, -0.5f, 0.5f, tot);
        tot = (H / 16) * (H / 32) * 256;
        pack_weights_kernel<<<(tot + 255) / 256, 256, 0, stream>>>(
            w2, w2p, H, H, H, 0.001f, 1.0f, tot);
        pack_weights_kernel<<<(tot + 255) / 256, 256, 0, stream>>>(
            w3, w3p, H, H, H, 0.001f, 1.0f, tot);
        tot = (OUTP / 16) * (H / 32) * 256;
        pack_weights_kernel<<<(tot + 255) / 256, 256, 0, stream>>>(
            w4, w4p, OUT, H, OUTP, 0.001f, 1.0f, tot);
    }
    // ---- x -> [T,B,IN] f16, /15 ----
    {
        int tot = T * B * IN;
        convert_x_kernel<<<(tot + 255) / 256, 256, 0, stream>>>(x, xh, T, B, IN);
    }
    // ---- zero membranes (harness poisons ws; must re-init every call) ----
    {
        size_t n = (size_t)3 * B * H + (size_t)B * OUTP;
        zero_f32_kernel<<<(unsigned)((n + 255) / 256), 256, 0, stream>>>(mems, n);
    }

    const size_t o2 = (size_t)T * B * H;
    const size_t o3 = 2 * o2;
    const size_t o4 = 3 * o2;
    const size_t o5 = o4 + (size_t)T * B * OUT;

    // tiles: layers 1-3: (B/16)*(H/16)=1024 tiles, 8 waves/block -> 128 blocks
    //        layer 4:    (B/16)*(OUTP/16)=48 tiles             -> 6 blocks
    for (int t = 0; t < T; ++t) {
        snn_layer_kernel<<<128, 256, 0, stream>>>(
            xh + (size_t)t * B * IN, w1p, mem1,
            out + (size_t)t * B * H, nullptr, act1,
            IN, H, H, 0.9f, 1.0f, 0);
        snn_layer_kernel<<<128, 256, 0, stream>>>(
            act1, w2p, mem2,
            out + o2 + (size_t)t * B * H, nullptr, act2,
            H, H, H, 0.85f, 1.0f, 0);
        snn_layer_kernel<<<128, 256, 0, stream>>>(
            act2, w3p, mem3,
            out + o3 + (size_t)t * B * H, nullptr, act3,
            H, H, H, 0.8f, 1.0f, 0);
        snn_layer_kernel<<<6, 256, 0, stream>>>(
            act3, w4p, mem4,
            out + o4 + (size_t)t * B * OUT, out + o5 + (size_t)t * B * OUT, nullptr,
            H, OUTP, OUT, 0.95f, 1.0f, 1);
    }
}
